// GAT_29437705847433
// MI455X (gfx1250) — matrix operand order repaired and checked by reading, compiled, not run on hardware
//
#include <hip/hip_runtime.h>
#include <hip/hip_bf16.h>

// Problem constants (match reference)
#define NN   50000      // nodes
#define NE   800000     // edges
#define NB   512        // graphs
#define FIN  128
#define NH   4          // heads
#define DH   64         // dim per head
#define HDIM 256        // NH*DH
#define NGF  200
#define GHF  256
#define CDIV(a,b) (((a)+(b)-1)/(b))

typedef __attribute__((ext_vector_type(8)))  _Float16 v8h;
typedef __attribute__((ext_vector_type(16))) _Float16 v16h;
typedef __attribute__((ext_vector_type(8)))  float    v8f;

// ---------------------------------------------------------------------------
// f32 -> f16 conversion with K padding (row-major). grid = (CDIV(Kp,256), M)
// ---------------------------------------------------------------------------
__global__ __launch_bounds__(256)
void k_cvt_a(const float* __restrict__ A, _Float16* __restrict__ A16, int K, int Kp)
{
    int k = blockIdx.x * blockDim.x + threadIdx.x;
    int m = blockIdx.y;
    if (k >= Kp) return;
    _Float16 v = (_Float16)0.f;
    if (k < K) v = (_Float16)A[(size_t)m * K + k];
    A16[(size_t)m * Kp + k] = v;
}

// f32 -> f16 transposed conversion with K and Nc padding: B16t[n*Kp+k] = B[k*Nc+n]
// grid = (CDIV(Kp,256), Ncp)
__global__ __launch_bounds__(256)
void k_cvt_bt(const float* __restrict__ B, _Float16* __restrict__ B16t,
              int K, int Kp, int Nc)
{
    int k = blockIdx.x * blockDim.x + threadIdx.x;
    int n = blockIdx.y;
    if (k >= Kp) return;
    _Float16 v = (_Float16)0.f;
    if (k < K && n < Nc) v = (_Float16)B[(size_t)k * Nc + n];
    B16t[(size_t)n * Kp + k] = v;
}

// ---------------------------------------------------------------------------
// WMMA GEMM: C[M,Nc] = A16[M,Kp] @ B16t[Ncp,Kp]^T (+bias)(+relu).
// One wave32 per 16x16 tile; unconditional aligned b128 fragment loads.
// M % 16 == 0, Kp % 32 == 0, Ncp % 16 == 0 guaranteed by caller.
// ---------------------------------------------------------------------------
__global__ __launch_bounds__(256)
void k_wmma_gemm(const _Float16* __restrict__ A16, const _Float16* __restrict__ B16t,
                 const float* __restrict__ bias, float* __restrict__ C,
                 int M, int Kp, int Nc, int Ncp, int relu)
{
    int gid  = blockIdx.x * blockDim.x + threadIdx.x;
    int wave = gid >> 5;
    int lane = threadIdx.x & 31;
    int tilesN = Ncp >> 4;
    int tilesM = M >> 4;
    if (wave >= tilesM * tilesN) return;        // wave-uniform exit (EXEC stays all-1)
    int tm = (wave / tilesN) << 4;
    int tn = (wave % tilesN) << 4;
    int half = lane >> 4;                        // 0 or 1
    int l15  = lane & 15;

    // A fragment (ISA layout): lane row = tm+l15, k = half*8 + {0..7, 16..23}
    const _Float16* Ap = A16 + (size_t)(tm + l15) * Kp + (half << 3);
    // B fragment: lane col = tn+l15, k = half*8 + {0..7, 16..23}
    const _Float16* Bp = B16t + (size_t)(tn + l15) * Kp + (half << 3);

    v8f acc = {};
    for (int k0 = 0; k0 < Kp; k0 += 32) {
        v8h a0 = *(const v8h*)(Ap + k0);
        v8h a1 = *(const v8h*)(Ap + k0 + 16);
        v8h b0 = *(const v8h*)(Bp + k0);
        v8h b1 = *(const v8h*)(Bp + k0 + 16);
        v16h a = __builtin_shufflevector(a0, a1, 0,1,2,3,4,5,6,7,8,9,10,11,12,13,14,15);
        v16h b = __builtin_shufflevector(b0, b1, 0,1,2,3,4,5,6,7,8,9,10,11,12,13,14,15);
        acc = __builtin_amdgcn_wmma_f32_16x16x32_f16(
                  false, a, false, b, (short)0, acc, false, false);
    }

    int bcol = tn + l15;
    if (bcol < Nc) {
        float bv = bias ? bias[bcol] : 0.f;
        #pragma unroll
        for (int j = 0; j < 8; ++j) {
            float v = acc[j] + bv;
            if (relu) v = fmaxf(v, 0.f);
            C[(size_t)(tm + (half << 3) + j) * Nc + bcol] = v;
        }
    }
}

// ---------------------------------------------------------------------------
// Per-node attention logits: el[n,h] = <f[n,h,:], al[h,:]>, er likewise.
// ---------------------------------------------------------------------------
__global__ __launch_bounds__(256)
void k_attn_logits(const float* __restrict__ f, const float* __restrict__ al,
                   const float* __restrict__ ar, float* __restrict__ el,
                   float* __restrict__ er, int nTot, int Dd)
{
    int i = blockIdx.x * blockDim.x + threadIdx.x;   // i = n*NH + h
    if (i >= nTot) return;
    int h = i & (NH - 1);
    const float* fr = f + (size_t)i * (size_t)Dd;
    float sl = 0.f, sr = 0.f;
    for (int d = 0; d < Dd; ++d) {
        float v = fr[d];
        sl += v * al[h * Dd + d];
        sr += v * ar[h * Dd + d];
    }
    el[i] = sl;
    er[i] = sr;
}

__global__ __launch_bounds__(256)
void k_fill(float* __restrict__ p, float v, int n)
{
    int i = blockIdx.x * blockDim.x + threadIdx.x;
    if (i < n) p[i] = v;
}

// float atomic max via sign-split int/uint atomics (init must be -inf/-1e30)
__device__ __forceinline__ void atomicMaxF(float* addr, float v)
{
    if (v >= 0.f) atomicMax((int*)addr, __float_as_int(v));
    else          atomicMin((unsigned int*)addr, __float_as_uint(v));
}

// Pass 1: e = leaky_relu(el[src]+er[dst]); store; segment max into mx[dst]
__global__ __launch_bounds__(256)
void k_edge_max(const int* __restrict__ src, const int* __restrict__ dst,
                const float* __restrict__ el, const float* __restrict__ er,
                float* __restrict__ ews, float* __restrict__ mx, int nE)
{
    int e = blockIdx.x * blockDim.x + threadIdx.x;
    if (e >= nE) return;
    int s = src[e], t = dst[e];
    #pragma unroll
    for (int h = 0; h < NH; ++h) {
        float v = el[s * NH + h] + er[t * NH + h];
        v = v > 0.f ? v : 0.2f * v;
        ews[(size_t)e * NH + h] = v;
        atomicMaxF(&mx[t * NH + h], v);
    }
}

// Pass 2: ex = exp(e - mx[dst]); store; segment sum into den[dst]
__global__ __launch_bounds__(256)
void k_edge_exp(const int* __restrict__ dst, float* __restrict__ ews,
                const float* __restrict__ mx, float* __restrict__ den, int nE)
{
    int e = blockIdx.x * blockDim.x + threadIdx.x;
    if (e >= nE) return;
    int t = dst[e];
    #pragma unroll
    for (int h = 0; h < NH; ++h) {
        float x = __expf(ews[(size_t)e * NH + h] - mx[t * NH + h]);
        ews[(size_t)e * NH + h] = x;
        atomicAdd(&den[t * NH + h], x);
    }
}

// Pass 3: rst[dst,h,d] += f[src,h,d] * (ex / den[dst,h]). One thread per (edge,d).
__global__ __launch_bounds__(256)
void k_edge_scatter(const int* __restrict__ src, const int* __restrict__ dst,
                    const float* __restrict__ ews, const float* __restrict__ den,
                    const float* __restrict__ f, float* __restrict__ rst,
                    int nE, int Dd, int sh)
{
    int i = blockIdx.x * blockDim.x + threadIdx.x;
    int e = i >> sh;
    int d = i & (Dd - 1);
    if (e >= nE) return;
    int s = src[e], t = dst[e];
    #pragma unroll
    for (int h = 0; h < NH; ++h) {
        float a = ews[(size_t)e * NH + h] / den[t * NH + h];
        atomicAdd(&rst[((size_t)t * NH + h) * Dd + d],
                  f[((size_t)s * NH + h) * Dd + d] * a);
    }
}

__global__ __launch_bounds__(256)
void k_elu(float* __restrict__ p, int n)
{
    int i = blockIdx.x * blockDim.x + threadIdx.x;
    if (i >= n) return;
    float v = p[i];
    p[i] = v > 0.f ? v : (__expf(v) - 1.f);
}

// Graph readout: latent[gid[n],h] += hn[n,h]  (C==1, mean over C is identity)
__global__ __launch_bounds__(256)
void k_readout(const float* __restrict__ hn, const int* __restrict__ gids,
               float* __restrict__ latent, int n)
{
    int i = blockIdx.x * blockDim.x + threadIdx.x;
    if (i >= n) return;
    int g = gids[i];
    #pragma unroll
    for (int h = 0; h < NH; ++h)
        atomicAdd(&latent[g * NH + h], hn[(size_t)i * NH + h]);
}

// Final: out[b] = lb + concat(latent[b,:], lg[b,:]) @ lw
__global__ __launch_bounds__(256)
void k_final(const float* __restrict__ latent, const float* __restrict__ lg,
             const float* __restrict__ lw, const float* __restrict__ lb,
             float* __restrict__ out, int nB)
{
    int b = blockIdx.x * blockDim.x + threadIdx.x;
    if (b >= nB) return;
    float acc = lb[0];
    #pragma unroll
    for (int h = 0; h < NH; ++h) acc += latent[b * NH + h] * lw[h];
    for (int j = 0; j < GHF; ++j) acc += lg[(size_t)b * GHF + j] * lw[NH + j];
    out[b] = acc;
}

// ---------------------------------------------------------------------------
static void gemm(const float* A, const float* B, const float* bias, float* C,
                 int M, int K, int Nc, int relu,
                 _Float16* A16, _Float16* B16, int cvtA, hipStream_t s)
{
    int Kp  = CDIV(K, 32) * 32;
    int Ncp = CDIV(Nc, 16) * 16;
    if (cvtA)
        k_cvt_a<<<dim3(CDIV(Kp, 256), M), 256, 0, s>>>(A, A16, K, Kp);
    k_cvt_bt<<<dim3(CDIV(Kp, 256), Ncp), 256, 0, s>>>(B, B16, K, Kp, Nc);
    int tiles = (M >> 4) * (Ncp >> 4);
    k_wmma_gemm<<<CDIV(tiles, 8), 256, 0, s>>>(A16, B16, bias, C, M, Kp, Nc, Ncp, relu);
}

static void fill(float* p, float v, int n, hipStream_t s)
{
    k_fill<<<CDIV(n, 256), 256, 0, s>>>(p, v, n);
}

extern "C" void kernel_launch(void* const* d_in, const int* in_sizes, int n_in,
                              void* d_out, int out_size, void* d_ws, size_t ws_size,
                              hipStream_t stream)
{
    const float* feat   = (const float*)d_in[0];
    const int*   src    = (const int*)  d_in[1];
    const int*   dst    = (const int*)  d_in[2];
    const int*   gids   = (const int*)  d_in[3];
    const float* g_feat = (const float*)d_in[4];
    const float* W0  = (const float*)d_in[5];
    const float* al0 = (const float*)d_in[6];
    const float* ar0 = (const float*)d_in[7];
    const float* W1  = (const float*)d_in[8];
    const float* al1 = (const float*)d_in[9];
    const float* ar1 = (const float*)d_in[10];
    const float* W2  = (const float*)d_in[11];
    const float* al2 = (const float*)d_in[12];
    const float* ar2 = (const float*)d_in[13];
    const float* W3  = (const float*)d_in[14];
    const float* al3 = (const float*)d_in[15];
    const float* ar3 = (const float*)d_in[16];
    const float* resW3 = (const float*)d_in[17];
    const float* gw1 = (const float*)d_in[18];
    const float* gb1 = (const float*)d_in[19];
    const float* gw2 = (const float*)d_in[20];
    const float* gb2 = (const float*)d_in[21];
    const float* lw  = (const float*)d_in[22];
    const float* lb  = (const float*)d_in[23];
    float* out = (float*)d_out;

    // Workspace layout. All float blocks are multiples of 16 elems -> 64B aligned.
    float* ws = (float*)d_ws;
    float* bufA   = ws;                          // [NN, HDIM]  h / rst (fp32 master)
    float* bufB   = bufA + (size_t)NN * HDIM;    // [NN, HDIM]  f
    float* el     = bufB + (size_t)NN * HDIM;    // [NN, NH]
    float* er     = el   + (size_t)NN * NH;
    float* mx     = er   + (size_t)NN * NH;
    float* den    = mx   + (size_t)NN * NH;
    float* ews    = den  + (size_t)NN * NH;      // [NE, NH]
    float* f3     = ews  + (size_t)NE * NH;      // [NN, NH] (C=1)
    float* hn     = f3   + (size_t)NN * NH;      // [NN, NH]
    float* latent = hn   + (size_t)NN * NH;      // [NB, NH]
    float* lg1    = latent + (size_t)NB * NH;    // [NB, GHF]
    float* lg2    = lg1  + (size_t)NB * GHF;     // [NB, GHF]
    _Float16* A16 = (_Float16*)(lg2 + (size_t)NB * GHF); // [NN, 256] f16 staged A
    _Float16* B16 = A16 + (size_t)NN * HDIM;             // [256, 256] f16 staged B^T

    const int egrid  = CDIV(NE, 256);
    const int nhgrid = CDIV(NN * NH, 256);
    const int sgrid  = CDIV(NE * DH, 256);

    // ---------------- Layer 0: feat[NN,128] -> bufA ----------------
    gemm(feat, W0, nullptr, bufB, NN, FIN, HDIM, 0, A16, B16, 1, stream);
    k_attn_logits<<<nhgrid, 256, 0, stream>>>(bufB, al0, ar0, el, er, NN * NH, DH);
    fill(mx, -1e30f, NN * NH, stream);
    fill(den, 0.f, NN * NH, stream);
    fill(bufA, 0.f, NN * HDIM, stream);          // rst starts at 0 (no residual)
    k_edge_max<<<egrid, 256, 0, stream>>>(src, dst, el, er, ews, mx, NE);
    k_edge_exp<<<egrid, 256, 0, stream>>>(dst, ews, mx, den, NE);
    k_edge_scatter<<<sgrid, 256, 0, stream>>>(src, dst, ews, den, bufB, bufA, NE, DH, 6);
    k_elu<<<CDIV(NN * HDIM, 256), 256, 0, stream>>>(bufA, NN * HDIM);

    // ---------------- Layers 1,2: residual folded into accumulator ----------------
    const float* Wsl[2] = { W1, W2 };
    const float* als[2] = { al1, al2 };
    const float* ars[2] = { ar1, ar2 };
    for (int L = 0; L < 2; ++L) {
        gemm(bufA, Wsl[L], nullptr, bufB, NN, HDIM, HDIM, 0, A16, B16, 1, stream);
        k_attn_logits<<<nhgrid, 256, 0, stream>>>(bufB, als[L], ars[L], el, er, NN * NH, DH);
        fill(mx, -1e30f, NN * NH, stream);
        fill(den, 0.f, NN * NH, stream);
        // rst initialized to h (res_id): accumulate directly on top of bufA
        k_edge_max<<<egrid, 256, 0, stream>>>(src, dst, el, er, ews, mx, NE);
        k_edge_exp<<<egrid, 256, 0, stream>>>(dst, ews, mx, den, NE);
        k_edge_scatter<<<sgrid, 256, 0, stream>>>(src, dst, ews, den, bufB, bufA, NE, DH, 6);
        k_elu<<<CDIV(NN * HDIM, 256), 256, 0, stream>>>(bufA, NN * HDIM);
    }

    // ---------------- Layer 3: [NN,256] -> [NN,4], res_w, no act ----------------
    // Convert A (bufA) once; reuse f16 copy for both W3 and resW3 GEMMs.
    gemm(bufA, W3,    nullptr, f3, NN, HDIM, NH, 0, A16, B16, 1, stream);
    gemm(bufA, resW3, nullptr, hn, NN, HDIM, NH, 0, A16, B16, 0, stream);
    k_attn_logits<<<nhgrid, 256, 0, stream>>>(f3, al3, ar3, el, er, NN * NH, 1);
    fill(mx, -1e30f, NN * NH, stream);
    fill(den, 0.f, NN * NH, stream);
    k_edge_max<<<egrid, 256, 0, stream>>>(src, dst, el, er, ews, mx, NE);
    k_edge_exp<<<egrid, 256, 0, stream>>>(dst, ews, mx, den, NE);
    k_edge_scatter<<<egrid, 256, 0, stream>>>(src, dst, ews, den, f3, hn, NE, 1, 0);

    // ---------------- Readout + graph MLP + final ----------------
    fill(latent, 0.f, NB * NH, stream);
    k_readout<<<CDIV(NN, 256), 256, 0, stream>>>(hn, gids, latent, NN);

    gemm(g_feat, gw1, gb1, lg1, NB, NGF, GHF, 1, A16, B16, 1, stream);  // relu(g@gw1+gb1)
    gemm(lg1,    gw2, gb2, lg2, NB, GHF, GHF, 1, A16, B16, 1, stream);  // relu(..@gw2+gb2)

    k_final<<<CDIV(NB, 256), 256, 0, stream>>>(latent, lg2, lw, lb, out, NB);
    (void)in_sizes; (void)n_in; (void)out_size; (void)ws_size;
}
